// GCN_20332375179669
// MI455X (gfx1250) — matrix operand order repaired and checked by reading, not compiled, not run on hardware
//
#include <hip/hip_runtime.h>
#include <hip/hip_bf16.h>

typedef __attribute__((ext_vector_type(16))) _Float16 v16h;
typedef __attribute__((ext_vector_type(8)))  float    v8f;
typedef __attribute__((ext_vector_type(4)))  unsigned int u32x4;
typedef __attribute__((ext_vector_type(8)))  int i32x8;
typedef __attribute__((ext_vector_type(4)))  int i32x4;

#define WMMA_F32_F16(A,B,C) \
  __builtin_amdgcn_wmma_f32_16x16x32_f16(false,(A),false,(B),(short)0,(C),false,false)

#define NB 8
#define NN 2048
#define HID 256
#define ATN_ 64
#define NNODE (NB*NN)
#define RPW 16      // query rows per wave
#define WPB 4       // waves per block
#define RPB 64      // query rows per block
#define CHUNK 32    // keys per chunk

#if defined(__gfx1250__) && __has_builtin(__builtin_amdgcn_tensor_load_to_lds) && \
    __has_builtin(__builtin_amdgcn_s_wait_tensorcnt)
#define HAVE_TDM 1
#else
#define HAVE_TDM 0
#endif

// A-fragment (16x32 f16): lane holds row M=lane&15.
// lanes 0-15: K = kk+0..15 (elems 0-15); lanes 16-31: K = kk+16..31
__device__ inline v16h load_a_frag16(const _Float16* __restrict__ row, int kk, int lane) {
  const int o0 = kk + ((lane & 16) ? 16 : 0);
  const int o1 = o0 + 8;
  v16h a;
#pragma unroll
  for (int i = 0; i < 8; ++i) { a[i] = row[o0 + i]; a[8 + i] = row[o1 + i]; }
  return a;
}

// B-fragment (32x16 f16): lane holds col N=lane&15 (colp = base + col*stride).
__device__ inline v16h load_b_frag16(const _Float16* __restrict__ colp, int kk, int lane) {
  const int o = kk + ((lane & 16) ? 16 : 0);
  v16h b;
#pragma unroll
  for (int i = 0; i < 16; ++i) b[i] = colp[o + i];
  return b;
}

#if HAVE_TDM
// Issue a 2-D TDM load: tile (tile_d0 x tile_d1) of f16 from a row-major tensor
// with innermost length tensor_d0 / stride stride0 (all in 2-byte elements),
// packed contiguously into LDS at lds_addr.
__device__ inline void tdm_load_2d(unsigned lds_addr, unsigned long long gaddr,
                                   unsigned tile_d0, unsigned tile_d1,
                                   unsigned tensor_d0, unsigned tensor_d1,
                                   unsigned long long stride0) {
  u32x4 g0 = {
    1u,                                                // count=1, user mode
    lds_addr,                                          // LDS byte address
    (unsigned)(gaddr & 0xFFFFFFFFull),                 // global addr lo
    (unsigned)((gaddr >> 32) & 0x1FFFFFFu) | (2u<<30)  // global addr hi | type=2
  };
  i32x8 g1 = {
    (int)(1u << 16),                                           // data_size=1 (2 bytes)
    (int)((tensor_d0 & 0xFFFFu) << 16),                        // tensor_dim0[15:0]
    (int)((tensor_d0 >> 16) | ((tensor_d1 & 0xFFFFu) << 16)),  // td0 hi | td1 lo
    (int)((tensor_d1 >> 16) | (tile_d0 << 16)),                // td1 hi | tile_dim0
    (int)(tile_d1 & 0xFFFFu),                                  // tile_dim1 (tile_dim2=0)
    (int)(unsigned)(stride0 & 0xFFFFFFFFull),                  // dim0 stride lo
    (int)(unsigned)((stride0 >> 32) & 0xFFFFu),                // dim0 stride hi
    0
  };
  i32x4 z4 = {0, 0, 0, 0};
#if __clang_major__ >= 23
  i32x8 z8 = {0, 0, 0, 0, 0, 0, 0, 0};
  __builtin_amdgcn_tensor_load_to_lds(g0, g1, z4, z4, z8, 0);
#else
  __builtin_amdgcn_tensor_load_to_lds(g0, g1, z4, z4, 0);
#endif
}
#endif

// Stage a 32-key chunk of K ([32][64] f16 -> kb) and V^T ([256][32] f16 -> vb).
__device__ inline void stage_chunk(const _Float16* __restrict__ kmat_b,
                                   const _Float16* __restrict__ vT_b, int j0,
                                   _Float16* kb, _Float16* vb, int tid, int wave) {
#if HAVE_TDM
  (void)tid;
  if (wave == 0) {
    tdm_load_2d((unsigned)(size_t)kb,
                (unsigned long long)(size_t)(kmat_b + (size_t)j0 * ATN_),
                /*tile*/ ATN_, CHUNK, /*tensor*/ ATN_, NN, /*stride0*/ ATN_);
    tdm_load_2d((unsigned)(size_t)vb,
                (unsigned long long)(size_t)(vT_b + j0),
                /*tile*/ CHUNK, HID, /*tensor*/ NN, HID, /*stride0*/ NN);
  }
#else
  (void)wave;
  // K: 32 keys x 64 halves = 4KB = 256 x 16B segments, 2 per thread
#pragma unroll
  for (int it = 0; it < 2; ++it) {
    const int e = it * 128 + tid;
    const int key = e >> 3, seg = e & 7;
    const int4 d = *(const int4*)(kmat_b + (size_t)(j0 + key) * ATN_ + seg * 8);
    *(int4*)(kb + key * ATN_ + seg * 8) = d;
  }
  // V: 256 cols x 32 keys = 16KB = 1024 x 16B segments, 8 per thread
#pragma unroll
  for (int it = 0; it < 8; ++it) {
    const int e = it * 128 + tid;
    const int c = e >> 2, quad = e & 3;
    const int4 d = *(const int4*)(vT_b + (size_t)c * NN + j0 + quad * 8);
    *(int4*)(vb + c * CHUNK + quad * 8) = d;
  }
#endif
}

__device__ inline void stage_wait(int wave) {
#if HAVE_TDM
  if (wave == 0) __builtin_amdgcn_s_wait_tensorcnt(0);
#endif
  __syncthreads();
}

// ---------------------------------------------------------------- weights -> f16, transposed
__global__ void k_cvt_weights(const float* __restrict__ Wh, const float* __restrict__ Wq,
                              const float* __restrict__ Wk, const float* __restrict__ Wv,
                              _Float16* __restrict__ WhT, _Float16* __restrict__ WqT,
                              _Float16* __restrict__ WkT, _Float16* __restrict__ WvT) {
  const int i = blockIdx.x * blockDim.x + threadIdx.x;
  const int cN = i >> 8;
  const int cK = i & 255;
  if (i < 256 * 256) {
    WhT[i] = (_Float16)Wh[cK * 256 + cN];
    WvT[i] = (_Float16)Wv[cK * 256 + cN];
  }
  if (i < 64 * 256) {
    WqT[i] = (_Float16)Wq[cK * 64 + cN];
    WkT[i] = (_Float16)Wk[cK * 64 + cN];
  }
}

// ---------------------------------------------------------------- embed + h0 = relu(feats@Wh + bh)
__global__ __launch_bounds__(32) void k_embed_h0(
    const int* __restrict__ x, const int* __restrict__ cue,
    const float* __restrict__ emb, const _Float16* __restrict__ WhT,
    const float* __restrict__ bh,
    float* __restrict__ h, _Float16* __restrict__ h16) {
  const int lane = threadIdx.x;
  const int mt = blockIdx.y, nt = blockIdx.x;
  const int node = mt * 16 + (lane & 15);
  const int xv = x[node];
  const float cuev = (float)cue[node];
  const float* erow = emb + (size_t)xv * 255;
  const int col = nt * 16 + (lane & 15);
  const _Float16* wcol = WhT + (size_t)col * 256;

  v8f c;
#pragma unroll
  for (int i = 0; i < 8; ++i) c[i] = 0.f;

#pragma unroll
  for (int kt = 0; kt < 8; ++kt) {
    const int kk = kt * 32;
    const int o0 = kk + ((lane & 16) ? 16 : 0);
    v16h a;
#pragma unroll
    for (int i = 0; i < 8; ++i) {
      const int k0 = o0 + i;      a[i]     = (_Float16)((k0 < 255) ? erow[k0] : cuev);
      const int k1 = o0 + 8 + i; a[8 + i] = (_Float16)((k1 < 255) ? erow[k1] : cuev);
    }
    v16h bf = load_b_frag16(wcol, kk, lane);
    c = WMMA_F32_F16(a, bf, c);
  }

  const float bias = bh[col];
  const int rbase = mt * 16 + ((lane & 16) ? 8 : 0);
#pragma unroll
  for (int r = 0; r < 8; ++r) {
    float v = fmaxf(c[r] + bias, 0.f);
    const size_t idx = (size_t)(rbase + r) * 256 + col;
    h[idx] = v;
    h16[idx] = (_Float16)v;
  }
}

// ---------------------------------------------------------------- q|k|v GEMM (v stored transposed)
__global__ __launch_bounds__(32) void k_qkv(
    const _Float16* __restrict__ h16,
    const _Float16* __restrict__ WqT, const _Float16* __restrict__ WkT,
    const _Float16* __restrict__ WvT,
    _Float16* __restrict__ q, _Float16* __restrict__ kout, _Float16* __restrict__ vT) {
  const int lane = threadIdx.x;
  const int mt = blockIdx.y, nt = blockIdx.x;    // nt: 0..23 (0-3 q, 4-7 k, 8-23 v)
  const int cg = nt * 16 + (lane & 15);
  const _Float16* wcol;
  if (cg < 64)        wcol = WqT + (size_t)cg * 256;
  else if (cg < 128)  wcol = WkT + (size_t)(cg - 64) * 256;
  else                wcol = WvT + (size_t)(cg - 128) * 256;
  const _Float16* arow = h16 + (size_t)(mt * 16 + (lane & 15)) * 256;

  v8f c;
#pragma unroll
  for (int i = 0; i < 8; ++i) c[i] = 0.f;
#pragma unroll
  for (int kt = 0; kt < 8; ++kt) {
    v16h a = load_a_frag16(arow, kt * 32, lane);
    v16h b = load_b_frag16(wcol, kt * 32, lane);
    c = WMMA_F32_F16(a, b, c);
  }

  const int rbase = mt * 16 + ((lane & 16) ? 8 : 0);
#pragma unroll
  for (int r = 0; r < 8; ++r) {
    const int row = rbase + r;
    const _Float16 val = (_Float16)c[r];
    if (cg < 64)       q[(size_t)row * 64 + cg] = val;
    else if (cg < 128) kout[(size_t)row * 64 + (cg - 64)] = val;
    else {
      const int bidx = row >> 11, nin = row & (NN - 1);
      vT[(size_t)bidx * HID * NN + (size_t)(cg - 128) * NN + nin] = val;
    }
  }
}

// ---------------------------------------------------------------- flash attention, split-K partials
// Block = 4 waves x 16 query rows; K/V chunks staged in LDS (TDM if available),
// double-buffered and shared by all 4 waves. Each split writes (O, m, l, hn) partials.
__global__ __launch_bounds__(128) void k_attn(
    const float* __restrict__ adj,
    const _Float16* __restrict__ q, const _Float16* __restrict__ kmat,
    const _Float16* __restrict__ vT, const int kps,
    float* __restrict__ Opart, float* __restrict__ mpart,
    float* __restrict__ lpart, float* __restrict__ hnpart) {
  __shared__ _Float16 kbuf[2][CHUNK * ATN_];   // 2 x 4KB
  __shared__ _Float16 vbuf[2][HID * CHUNK];    // 2 x 16KB
  __shared__ _Float16 pbuf[WPB][16 * 32];      // 4 x 1KB

  const int tid = (int)threadIdx.x;
  const int lane = tid & 31;
  const int wave = tid >> 5;
  const int b = blockIdx.y;
  const int sidx = blockIdx.z;
  const int i0 = blockIdx.x * RPB + wave * RPW;
  const int nodebase = b * NN;
  const int col = lane & 15;
  const int rowoff = (lane & 16) ? 8 : 0;
  const int jbeg = sidx * kps;
  const int jend = jbeg + kps;

  const _Float16* kmat_b = kmat + (size_t)nodebase * ATN_;
  const _Float16* vT_b = vT + (size_t)b * HID * NN;
  const float* adjbase = adj + ((size_t)b * NN + i0) * NN;

  const _Float16* qrow = q + (size_t)(nodebase + i0 + col) * ATN_;
  const v16h qa0 = load_a_frag16(qrow, 0, lane);
  const v16h qa1 = load_a_frag16(qrow, 32, lane);

  v8f O[16];
#pragma unroll
  for (int t = 0; t < 16; ++t)
#pragma unroll
    for (int i = 0; i < 8; ++i) O[t][i] = 0.f;

  float m[8], l[8], hn[8];
#pragma unroll
  for (int r = 0; r < 8; ++r) { m[r] = -1e30f; l[r] = 0.f; hn[r] = 0.f; }

  stage_chunk(kmat_b, vT_b, jbeg, kbuf[0], vbuf[0], tid, wave);
  stage_wait(wave);

  int cur = 0;
  for (int j0 = jbeg; j0 < jend; j0 += CHUNK) {
    const int nxt = cur ^ 1;
    if (j0 + CHUNK < jend) {
      stage_chunk(kmat_b, vT_b, j0 + CHUNK, kbuf[nxt], vbuf[nxt], tid, wave);
      __builtin_prefetch(adjbase + j0 + CHUNK, 0, 1);
    }

    // S = Q @ K^T from LDS (two 16x16 tiles over 32 keys, K-dim 64)
    v8f s0, s1;
#pragma unroll
    for (int i = 0; i < 8; ++i) { s0[i] = 0.f; s1[i] = 0.f; }
    {
      const int hh = (lane & 16) ? 16 : 0;
      v16h b0a, b0b, b1a, b1b;
#pragma unroll
      for (int i = 0; i < 16; ++i) {
        b0a[i] = kbuf[cur][col * ATN_ + hh + i];
        b0b[i] = kbuf[cur][col * ATN_ + 32 + hh + i];
        b1a[i] = kbuf[cur][(16 + col) * ATN_ + hh + i];
        b1b[i] = kbuf[cur][(16 + col) * ATN_ + 32 + hh + i];
      }
      s0 = WMMA_F32_F16(qa0, b0a, s0);
      s0 = WMMA_F32_F16(qa1, b0b, s0);
      s1 = WMMA_F32_F16(qa0, b1a, s1);
      s1 = WMMA_F32_F16(qa1, b1b, s1);
    }

    // adj mask + online softmax (row r across this lane's 16-lane group)
    float scale[8];
#pragma unroll
    for (int r = 0; r < 8; ++r) {
      const float* arow = adjbase + (size_t)(r + rowoff) * NN + j0;
      const float a0 = arow[col];
      const float a1 = arow[16 + col];
      const float v0 = (a0 > 0.f) ? s0[r] : -1e9f;
      const float v1 = (a1 > 0.f) ? s1[r] : -1e9f;
      hn[r] = fmaxf(hn[r], fmaxf((a0 > 0.f) ? 1.f : 0.f, (a1 > 0.f) ? 1.f : 0.f));
      float rmax = fmaxf(v0, v1);
#pragma unroll
      for (int d = 1; d < 16; d <<= 1) rmax = fmaxf(rmax, __shfl_xor(rmax, d, 16));
      const float mn = fmaxf(m[r], rmax);
      const float sc = __expf(m[r] - mn);
      const float p0 = __expf(v0 - mn);
      const float p1 = __expf(v1 - mn);
      s0[r] = p0; s1[r] = p1;
      float rs = p0 + p1;
#pragma unroll
      for (int d = 1; d < 16; d <<= 1) rs += __shfl_xor(rs, d, 16);
      l[r] = l[r] * sc + rs;
      m[r] = mn;
      scale[r] = sc;
    }
#pragma unroll
    for (int t = 0; t < 16; ++t)
#pragma unroll
      for (int r = 0; r < 8; ++r) O[t][r] *= scale[r];

    // transpose P (C-layout f32) -> f16 A-fragment via per-wave LDS region
    // (same-wave DS ops are in order; no cross-wave sharing of pbuf)
#pragma unroll
    for (int r = 0; r < 8; ++r) {
      const int row = r + rowoff;
      pbuf[wave][row * 32 + col]      = (_Float16)s0[r];
      pbuf[wave][row * 32 + 16 + col] = (_Float16)s1[r];
    }
    v16h pA;
    {
      const int o0 = (lane & 16) ? 16 : 0;
#pragma unroll
      for (int i = 0; i < 8; ++i) {
        pA[i]     = pbuf[wave][col * 32 + o0 + i];
        pA[8 + i] = pbuf[wave][col * 32 + o0 + 8 + i];
      }
    }

    // O += P @ V from LDS (16 output tiles across HIDDEN)
    {
      const int hh = (lane & 16) ? 16 : 0;
#pragma unroll
      for (int t = 0; t < 16; ++t) {
        v16h vB;
#pragma unroll
        for (int i = 0; i < 16; ++i) vB[i] = vbuf[cur][(t * 16 + col) * CHUNK + hh + i];
        O[t] = WMMA_F32_F16(pA, vB, O[t]);
      }
    }

    stage_wait(wave);  // next chunk complete + all waves done reading cur
    cur = nxt;
  }

  // write partials
#pragma unroll
  for (int r = 0; r < 8; ++r) {
#pragma unroll
    for (int d = 1; d < 16; d <<= 1) hn[r] = fmaxf(hn[r], __shfl_xor(hn[r], d, 16));
  }
  const size_t pbase = (size_t)sidx * NNODE + nodebase + i0;
  if (col == 0) {
#pragma unroll
    for (int r = 0; r < 8; ++r) {
      mpart[pbase + r + rowoff] = m[r];
      lpart[pbase + r + rowoff] = l[r];
      hnpart[pbase + r + rowoff] = hn[r];
    }
  }
#pragma unroll
  for (int t = 0; t < 16; ++t)
#pragma unroll
    for (int r = 0; r < 8; ++r)
      Opart[(pbase + r + rowoff) * HID + t * 16 + col] = O[t][r];
}

// ---------------------------------------------------------------- combine splits + bias + sigmoid
__global__ void k_attn_combine(const float* __restrict__ Opart, const float* __restrict__ mpart,
                               const float* __restrict__ lpart, const float* __restrict__ hnpart,
                               const float* __restrict__ gcb, const int nsplit,
                               float* __restrict__ h, _Float16* __restrict__ h16) {
  const int idx = blockIdx.x * blockDim.x + threadIdx.x;
  if (idx >= NNODE * HID) return;
  const int node = idx >> 8, cg = idx & 255;
  float mx = -1e30f, hn = 0.f;
  for (int s = 0; s < nsplit; ++s) {
    mx = fmaxf(mx, mpart[(size_t)s * NNODE + node]);
    hn = fmaxf(hn, hnpart[(size_t)s * NNODE + node]);
  }
  float lsum = 0.f, osum = 0.f;
  for (int s = 0; s < nsplit; ++s) {
    const float w = __expf(mpart[(size_t)s * NNODE + node] - mx);
    lsum += lpart[(size_t)s * NNODE + node] * w;
    osum += Opart[((size_t)s * NNODE + node) * HID + cg] * w;
  }
  float o = (hn > 0.f) ? (osum / lsum) : 0.f;
  o += gcb[cg];
  const float sig = 1.f / (1.f + __expf(-o));
  h[idx] = sig;
  h16[idx] = (_Float16)sig;
}

// ---------------------------------------------------------------- classifier + softmax(2)
__global__ void k_cls(const float* __restrict__ h, const float* __restrict__ Wc,
                      const float* __restrict__ bc, float* __restrict__ out) {
  const int node = blockIdx.x * blockDim.x + threadIdx.x;
  if (node >= NNODE) return;
  const float* hr = h + (size_t)node * 256;
  float l0 = bc[0], l1 = bc[1];
#pragma unroll 4
  for (int kidx = 0; kidx < 256; ++kidx) {
    const float hv = hr[kidx];
    l0 = fmaf(hv, Wc[kidx * 2 + 0], l0);
    l1 = fmaf(hv, Wc[kidx * 2 + 1], l1);
  }
  const float mx = fmaxf(l0, l1);
  const float e0 = __expf(l0 - mx), e1 = __expf(l1 - mx);
  const float inv = 1.f / (e0 + e1);
  out[(size_t)node * 2 + 0] = e0 * inv;
  out[(size_t)node * 2 + 1] = e1 * inv;
}

// ---------------------------------------------------------------- launch
extern "C" void kernel_launch(void* const* d_in, const int* in_sizes, int n_in,
                              void* d_out, int out_size, void* d_ws, size_t ws_size,
                              hipStream_t stream) {
  const int*   x    = (const int*)d_in[0];
  const int*   cue  = (const int*)d_in[1];
  const float* adj  = (const float*)d_in[2];
  const float* emb  = (const float*)d_in[3];
  const float* Wh   = (const float*)d_in[4];
  const float* bh   = (const float*)d_in[5];
  const float* Wq   = (const float*)d_in[6];
  const float* Wk   = (const float*)d_in[7];
  const float* Wv   = (const float*)d_in[8];
  const float* gcb  = (const float*)d_in[9];
  const float* Wc   = (const float*)d_in[10];
  const float* bc   = (const float*)d_in[11];
  float* out = (float*)d_out;
  (void)in_sizes; (void)n_in; (void)out_size;

  char* ws = (char*)d_ws;
  size_t off = 0;
  _Float16* WhT = (_Float16*)(ws + off); off += (size_t)256 * 256 * 2;
  _Float16* WqT = (_Float16*)(ws + off); off += (size_t)64 * 256 * 2;
  _Float16* WkT = (_Float16*)(ws + off); off += (size_t)64 * 256 * 2;
  _Float16* WvT = (_Float16*)(ws + off); off += (size_t)256 * 256 * 2;
  float*    h   = (float*)(ws + off);    off += (size_t)NNODE * 256 * 4;
  _Float16* h16 = (_Float16*)(ws + off); off += (size_t)NNODE * 256 * 2;
  _Float16* qb  = (_Float16*)(ws + off); off += (size_t)NNODE * 64 * 2;
  _Float16* kb  = (_Float16*)(ws + off); off += (size_t)NNODE * 64 * 2;
  _Float16* vT  = (_Float16*)(ws + off); off += (size_t)NB * HID * NN * 2;

  // split-K factor chosen from available workspace (deterministic: ws_size fixed)
  auto partial_need = [](int S) {
    return (size_t)S * NNODE * HID * 4 + 3ull * (size_t)S * NNODE * 4;
  };
  int S = 1;
  if (ws_size >= off + partial_need(4)) S = 4;
  else if (ws_size >= off + partial_need(2)) S = 2;
  float* Opart  = (float*)(ws + off); off += (size_t)S * NNODE * HID * 4;
  float* mpart  = (float*)(ws + off); off += (size_t)S * NNODE * 4;
  float* lpart  = (float*)(ws + off); off += (size_t)S * NNODE * 4;
  float* hnpart = (float*)(ws + off); off += (size_t)S * NNODE * 4;
  const int kps = NN / S;

  k_cvt_weights<<<dim3(256), dim3(256), 0, stream>>>(Wh, Wq, Wk, Wv, WhT, WqT, WkT, WvT);
  k_embed_h0<<<dim3(16, 1024), dim3(32), 0, stream>>>(x, cue, emb, WhT, bh, h, h16);
  for (int layer = 0; layer < 2; ++layer) {
    k_qkv<<<dim3(24, 1024), dim3(32), 0, stream>>>(h16, WqT, WkT, WvT, qb, kb, vT);
    k_attn<<<dim3(NN / RPB, NB, S), dim3(128), 0, stream>>>(
        adj, qb, kb, vT, kps, Opart, mpart, lpart, hnpart);
    k_attn_combine<<<dim3((NNODE * HID) / 256), dim3(256), 0, stream>>>(
        Opart, mpart, lpart, hnpart, gcb, S, h, h16);
  }
  k_cls<<<dim3(64), dim3(256), 0, stream>>>(h, Wc, bc, out);
}
